// DotProductAttention_11089605558274
// MI455X (gfx1250) — compile-verified, matrix operand order checked by reading, not run on hardware
//
#include <hip/hip_runtime.h>

// CDNA5 (gfx1250) fused flash-attention:  O = softmax(mask(Q K^T / sqrt(d))) V
// B=8, Lq=Lk=2048, d=64, fp32 in/out, f16 WMMA with f32 accumulation.
// 64-key tiles, row-sum via P*ones WMMA, packed-f16 max butterflies,
// 16B-aligned padded LDS rows (b128 fragment loads), mask-specialized tiles.

typedef _Float16 half16 __attribute__((ext_vector_type(16)));
typedef _Float16 half2v __attribute__((ext_vector_type(2)));
typedef float    float8 __attribute__((ext_vector_type(8)));

#define NB     8
#define LQ     2048
#define LK     2048
#define DH     64
#define LOG2E  1.44269504f
#define QSCL   (0.125f * LOG2E)      // fold 1/sqrt(d) and log2(e) into Q
#define MASKV  (-60000.0f)           // exactly representable in f16; exp2 -> 0
#define LST    72                    // padded LDS row stride: 144B, 16B-aligned,
                                     // dword stride 36 -> 16 distinct banks

#define EXP2F(x) __builtin_amdgcn_exp2f(x)   // single v_exp_f32

// A/B fragment K-index map for V_WMMA_*_16X16X32_F16 (ISA 7.12.2):
// VGPR v (f16 pair) holds K = kb[v]+8*halfgroup, kb[v]+1+8*halfgroup.
// kb dwords {0..3} and {8..11} are contiguous -> two ds_load_b128 per fragment.
__device__ __forceinline__ half16 ld_frag_lds(const _Float16* p, int hoff) {
  const int kb[8] = {0, 2, 4, 6, 16, 18, 20, 22};
  union { half16 h; unsigned u[8]; } f;
#pragma unroll
  for (int v = 0; v < 8; ++v)
    f.u[v] = *reinterpret_cast<const unsigned*>(p + kb[v] + hoff);
  return f.h;
}

__device__ __forceinline__ half16 ld_qfrag_global(const float* p, int hoff) {
  const int kb[8] = {0, 2, 4, 6, 16, 18, 20, 22};
  union { half16 h; unsigned u[8]; } f;
#pragma unroll
  for (int v = 0; v < 8; ++v) {
    union { _Float16 h2[2]; unsigned u; } pk;
    pk.h2[0] = (_Float16)(p[kb[v] + hoff]     * QSCL);
    pk.h2[1] = (_Float16)(p[kb[v] + hoff + 1] * QSCL);
    f.u[v] = pk.u;
  }
  return f.h;
}

__launch_bounds__(256)
__global__ void fa_fwd_kernel(const float* __restrict__ Q, const float* __restrict__ K,
                              const float* __restrict__ V, const int* __restrict__ VL,
                              float* __restrict__ Out) {
  __shared__ alignas(16) _Float16 Klds[64 * LST];      // [key][dim]
  __shared__ alignas(16) _Float16 Vlds[64 * LST];      // [dim][key] (transposed)
  __shared__ alignas(16) _Float16 Plds[8][16 * LST];   // per-wave P staging

  const int tid  = threadIdx.x;
  const int wave = tid >> 5;
  const int lane = tid & 31;
  const int hl   = lane >> 4;      // 16-lane half-group
  const int ln   = lane & 15;
  const int hoff = hl * 8;

  const int b   = blockIdx.x >> 4;        // batch
  const int qb  = blockIdx.x & 15;        // 128-row query block
  const int qr0 = qb * 128 + wave * 16;   // this wave's first query row

  int vlen = VL[b];
  vlen = vlen < 0 ? 0 : (vlen > LK ? LK : vlen);
  // Fully masked tail tiles contribute exactly 0 in fp32 -> skip.
  // vlen==0 degenerates to uniform softmax over ALL keys -> run all (masked).
  const int nfull = vlen >> 6;                                   // unmasked tiles
  const int nblk  = (vlen == 0) ? (LK / 64) : ((vlen + 63) >> 6); // total tiles

  // Q A-fragments (16x32 f16, pre-scaled), row m = ln.
  const float* qrow = Q + ((size_t)(b * LQ + qr0 + ln)) * DH;
  const half16 qa0 = ld_qfrag_global(qrow,      hoff);  // dims  0..31
  const half16 qa1 = ld_qfrag_global(qrow + 32, hoff);  // dims 32..63

  // all-ones B fragment for row-sum-via-WMMA (layout independent)
  half16 onesf;
#pragma unroll
  for (int j = 0; j < 16; ++j) onesf[j] = (_Float16)1.0f;

  float8 o0 = {}, o1 = {}, o2 = {}, o3 = {}, ol = {};  // O accums + row-sum accum
  float mi[8];
#pragma unroll
  for (int r = 0; r < 8; ++r) mi[r] = -3.0e38f;

  auto process_tile = [&](int blk, bool DOMASK) __attribute__((always_inline)) {
    __syncthreads();  // previous tile's fragment reads done before overwrite
    const size_t kvbase = ((size_t)(b * LK + blk * 64)) * DH;
#pragma unroll
    for (int i = 0; i < 8; ++i) {
      const int p = (i << 8) + tid;           // 0..2047 (pair index)
      // K tile: pair of consecutive dims, [key][dim] rows
      {
        const int ky = p >> 5, dm2 = (p & 31) << 1;
        const float2 kv = *reinterpret_cast<const float2*>(&K[kvbase + ky * DH + dm2]);
        union { half2v h; unsigned u; } pk;
        pk.h[0] = (_Float16)kv.x; pk.h[1] = (_Float16)kv.y;
        *reinterpret_cast<unsigned*>(&Klds[ky * LST + dm2]) = pk.u;
      }
      // V tile: pair of consecutive keys at same dim, transposed [dim][key]
      {
        const int dm = p & 63, ky2 = (p >> 6) << 1;
        union { half2v h; unsigned u; } pk;
        pk.h[0] = (_Float16)V[kvbase + ky2 * DH + dm];
        pk.h[1] = (_Float16)V[kvbase + (ky2 + 1) * DH + dm];
        *reinterpret_cast<unsigned*>(&Vlds[dm * LST + ky2]) = pk.u;
      }
      if (blk + 1 < nblk) {
        __builtin_prefetch(&K[kvbase + 64 * DH + (p << 1)], 0, 1);
        __builtin_prefetch(&V[kvbase + 64 * DH + (p << 1)], 0, 1);
      }
    }
    __syncthreads();

    // ---- S = Q K^T : four 16x16 score tiles (keys nt*16 .. nt*16+15) ----
    float8 s0 = {}, s1 = {}, s2 = {}, s3 = {};
    {
      const _Float16* kp0 = &Klds[(0 * 16 + ln) * LST];
      const _Float16* kp1 = &Klds[(1 * 16 + ln) * LST];
      const _Float16* kp2 = &Klds[(2 * 16 + ln) * LST];
      const _Float16* kp3 = &Klds[(3 * 16 + ln) * LST];
      s0 = __builtin_amdgcn_wmma_f32_16x16x32_f16(false, qa0, false, ld_frag_lds(kp0,      hoff), (short)0, s0, false, false);
      s0 = __builtin_amdgcn_wmma_f32_16x16x32_f16(false, qa1, false, ld_frag_lds(kp0 + 32, hoff), (short)0, s0, false, false);
      s1 = __builtin_amdgcn_wmma_f32_16x16x32_f16(false, qa0, false, ld_frag_lds(kp1,      hoff), (short)0, s1, false, false);
      s1 = __builtin_amdgcn_wmma_f32_16x16x32_f16(false, qa1, false, ld_frag_lds(kp1 + 32, hoff), (short)0, s1, false, false);
      s2 = __builtin_amdgcn_wmma_f32_16x16x32_f16(false, qa0, false, ld_frag_lds(kp2,      hoff), (short)0, s2, false, false);
      s2 = __builtin_amdgcn_wmma_f32_16x16x32_f16(false, qa1, false, ld_frag_lds(kp2 + 32, hoff), (short)0, s2, false, false);
      s3 = __builtin_amdgcn_wmma_f32_16x16x32_f16(false, qa0, false, ld_frag_lds(kp3,      hoff), (short)0, s3, false, false);
      s3 = __builtin_amdgcn_wmma_f32_16x16x32_f16(false, qa1, false, ld_frag_lds(kp3 + 32, hoff), (short)0, s3, false, false);
    }

    // ---- mask (only the boundary / all-masked tiles pay for cndmasks) ----
    float v0[8], v1[8], v2[8], v3[8], mx[8];
    if (DOMASK) {
      const int  kb0 = blk * 64 + ln;
      const bool m0 = (kb0      ) >= vlen;
      const bool m1 = (kb0 + 16 ) >= vlen;
      const bool m2 = (kb0 + 32 ) >= vlen;
      const bool m3 = (kb0 + 48 ) >= vlen;
#pragma unroll
      for (int r = 0; r < 8; ++r) {
        v0[r] = m0 ? MASKV : s0[r];
        v1[r] = m1 ? MASKV : s1[r];
        v2[r] = m2 ? MASKV : s2[r];
        v3[r] = m3 ? MASKV : s3[r];
      }
    } else {
#pragma unroll
      for (int r = 0; r < 8; ++r) { v0[r] = s0[r]; v1[r] = s1[r]; v2[r] = s2[r]; v3[r] = s3[r]; }
    }
#pragma unroll
    for (int r = 0; r < 8; ++r)
      mx[r] = fmaxf(fmaxf(v0[r], v1[r]), fmaxf(v2[r], v3[r]));

    // ---- per-row max across the 16-lane N group, packed as f16 pairs ----
    // (any consistent per-row upper bound is exact for softmax; f16 RNE is fine)
#pragma unroll
    for (int t = 0; t < 4; ++t) {
      union { half2v h; int i; } pk, ot;
      pk.h[0] = (_Float16)mx[2 * t];
      pk.h[1] = (_Float16)mx[2 * t + 1];
#pragma unroll
      for (int s = 8; s >= 1; s >>= 1) {
        ot.i = __shfl_xor(pk.i, s, 32);
        pk.h = __builtin_elementwise_max(pk.h, ot.h);
      }
      mx[2 * t]     = (float)pk.h[0];
      mx[2 * t + 1] = (float)pk.h[1];
    }

    // ---- online softmax update + stage P to LDS ----
    _Float16* prow = Plds[wave];
#pragma unroll
    for (int r = 0; r < 8; ++r) {
      const float mnew  = fmaxf(mi[r], mx[r]);
      const float alpha = EXP2F(mi[r] - mnew);
      mi[r] = mnew;
      o0[r] *= alpha; o1[r] *= alpha; o2[r] *= alpha; o3[r] *= alpha; ol[r] *= alpha;
      const float p0 = EXP2F(v0[r] - mnew);
      const float p1 = EXP2F(v1[r] - mnew);
      const float p2 = EXP2F(v2[r] - mnew);
      const float p3 = EXP2F(v3[r] - mnew);
      const int m = r + hoff;
      prow[m * LST + ln]      = (_Float16)p0;
      prow[m * LST + 16 + ln] = (_Float16)p1;
      prow[m * LST + 32 + ln] = (_Float16)p2;
      prow[m * LST + 48 + ln] = (_Float16)p3;
    }

    // wave-local LDS RAW: P writes above feed fragment reads below
    asm volatile("s_wait_dscnt 0" ::: "memory");

    // ---- O += P V ;  l += P * ones  (two P A-fragments: keys 0-31 / 32-63) ----
    const half16 pa0 = ld_frag_lds(&Plds[wave][ln * LST],      hoff);
    const half16 pa1 = ld_frag_lds(&Plds[wave][ln * LST + 32], hoff);
    const _Float16* vp0 = &Vlds[(0 * 16 + ln) * LST];
    const _Float16* vp1 = &Vlds[(1 * 16 + ln) * LST];
    const _Float16* vp2 = &Vlds[(2 * 16 + ln) * LST];
    const _Float16* vp3 = &Vlds[(3 * 16 + ln) * LST];
    o0 = __builtin_amdgcn_wmma_f32_16x16x32_f16(false, pa0, false, ld_frag_lds(vp0,      hoff), (short)0, o0, false, false);
    o1 = __builtin_amdgcn_wmma_f32_16x16x32_f16(false, pa0, false, ld_frag_lds(vp1,      hoff), (short)0, o1, false, false);
    o2 = __builtin_amdgcn_wmma_f32_16x16x32_f16(false, pa0, false, ld_frag_lds(vp2,      hoff), (short)0, o2, false, false);
    o3 = __builtin_amdgcn_wmma_f32_16x16x32_f16(false, pa0, false, ld_frag_lds(vp3,      hoff), (short)0, o3, false, false);
    ol = __builtin_amdgcn_wmma_f32_16x16x32_f16(false, pa0, false, onesf,                 (short)0, ol, false, false);
    o0 = __builtin_amdgcn_wmma_f32_16x16x32_f16(false, pa1, false, ld_frag_lds(vp0 + 32, hoff), (short)0, o0, false, false);
    o1 = __builtin_amdgcn_wmma_f32_16x16x32_f16(false, pa1, false, ld_frag_lds(vp1 + 32, hoff), (short)0, o1, false, false);
    o2 = __builtin_amdgcn_wmma_f32_16x16x32_f16(false, pa1, false, ld_frag_lds(vp2 + 32, hoff), (short)0, o2, false, false);
    o3 = __builtin_amdgcn_wmma_f32_16x16x32_f16(false, pa1, false, ld_frag_lds(vp3 + 32, hoff), (short)0, o3, false, false);
    ol = __builtin_amdgcn_wmma_f32_16x16x32_f16(false, pa1, false, onesf,                 (short)0, ol, false, false);
  };

  for (int blk = 0; blk < nfull; ++blk)  process_tile(blk, false);  // no masking
  for (int blk = nfull; blk < nblk; ++blk) process_tile(blk, true); // boundary / vlen==0

  // ---- epilogue: normalize by row-sum and store fp32 ----
#pragma unroll
  for (int r = 0; r < 8; ++r) {
    const float inv = 1.0f / ol[r];
    const int m = r + hoff;
    float* orow = Out + ((size_t)(b * LQ + qr0 + m)) * DH + ln;
    orow[0]  = o0[r] * inv;
    orow[16] = o1[r] * inv;
    orow[32] = o2[r] * inv;
    orow[48] = o3[r] * inv;
  }
}

extern "C" void kernel_launch(void* const* d_in, const int* in_sizes, int n_in,
                              void* d_out, int out_size, void* d_ws, size_t ws_size,
                              hipStream_t stream) {
  (void)in_sizes; (void)n_in; (void)out_size; (void)d_ws; (void)ws_size;
  const float* Q  = (const float*)d_in[0];
  const float* K  = (const float*)d_in[1];
  const float* V  = (const float*)d_in[2];
  const int*   VL = (const int*)d_in[3];
  float* Out = (float*)d_out;
  // 8 batches * 16 query blocks (128 rows each); 256 threads = 8 waves/block
  fa_fwd_kernel<<<dim3(NB * (LQ / 128)), dim3(256), 0, stream>>>(Q, K, V, VL, Out);
}
